// SlotPathE_44032004718737
// MI455X (gfx1250) — matrix operand order repaired and checked by reading, not compiled, not run on hardware
//
#include <hip/hip_runtime.h>

// ---------------------------------------------------------------------------
// SlotPathE forward for MI455X (gfx1250, wave32, WMMA).
// Heavy GEMMs: v_wmma_f32_16x16x32_bf16, f32 accumulation, double-buffered
// LDS, staged via GLOBAL_LOAD_ASYNC_TO_LDS_B128 when available.
// ---------------------------------------------------------------------------

typedef unsigned short u16;
typedef __attribute__((ext_vector_type(16))) __bf16 v16bf;
typedef __attribute__((ext_vector_type(8)))  __bf16 v8bf;
typedef __attribute__((ext_vector_type(8)))  float  v8f;

#define DM   1024     // d_model
#define SDIM 1152     // slot_dim
#define NSL  64       // num_slots
#define BATCH 4
#define TSEQ 2048
#define BT   (BATCH * TSEQ)
#define KTOP 8

#if __has_builtin(__builtin_amdgcn_global_load_async_to_lds_b128) && \
    __has_builtin(__builtin_amdgcn_s_wait_asynccnt)
#define USE_ASYNC 1
#else
#define USE_ASYNC 0
#endif

__device__ __forceinline__ u16 f2bf(float f) {
    unsigned u = __float_as_uint(f);
    u += 0x7fffu + ((u >> 16) & 1u);      // round-to-nearest-even
    return (u16)(u >> 16);
}
__device__ __forceinline__ float bf2f(u16 h) {
    return __uint_as_float(((unsigned)h) << 16);
}

#if USE_ASYNC
typedef int v4i_vs __attribute__((vector_size(16)));
typedef __attribute__((address_space(1))) v4i_vs* spe_gptr_t;
typedef __attribute__((address_space(3))) v4i_vs* spe_lptr_t;
__device__ __forceinline__ void async_b128(const u16* g, u16* l) {
    __builtin_amdgcn_global_load_async_to_lds_b128(
        (spe_gptr_t)(void*)(g),
        (spe_lptr_t)(void*)(l),
        0, 0);
}
#endif

// ---------------------------------------------------------------------------
// f32 -> bf16 row-major conversion (grid-stride)
// ---------------------------------------------------------------------------
__global__ __launch_bounds__(256) void spe_cvt_kernel(const float* __restrict__ s,
                                                      u16* __restrict__ d, int n) {
    for (int i = blockIdx.x * blockDim.x + threadIdx.x; i < n;
         i += gridDim.x * blockDim.x)
        d[i] = f2bf(s[i]);
}

// ---------------------------------------------------------------------------
// f32 [z][R][C] -> bf16 transposed [z][C][R], tiled through LDS.
// R, C multiples of 32. Block = 256 threads.
// ---------------------------------------------------------------------------
__global__ __launch_bounds__(256) void spe_cvtT_kernel(const float* __restrict__ src,
                                                       u16* __restrict__ dst,
                                                       int R, int Ccols,
                                                       long long sStride, long long dStride) {
    __shared__ u16 tile[32][33];
    const int z  = blockIdx.z;
    const int r0 = blockIdx.y << 5, c0 = blockIdx.x << 5;
    const int tx = threadIdx.x & 31, ty = threadIdx.x >> 5;
    const float* s = src + (size_t)z * sStride;
    u16* d = dst + (size_t)z * dStride;
    #pragma unroll
    for (int i = 0; i < 4; ++i)
        tile[ty + i * 8][tx] = f2bf(s[(size_t)(r0 + ty + i * 8) * Ccols + c0 + tx]);
    __syncthreads();
    #pragma unroll
    for (int i = 0; i < 4; ++i)
        d[(size_t)(c0 + ty + i * 8) * R + r0 + tx] = tile[tx][ty + i * 8];
}

// ---------------------------------------------------------------------------
// slot_mean[d] = mean_s slot_init[s, d], d < DM
// ---------------------------------------------------------------------------
__global__ __launch_bounds__(256) void spe_slotmean_kernel(const float* __restrict__ slot_init,
                                                           float* __restrict__ sm) {
    int d = blockIdx.x * blockDim.x + threadIdx.x;
    if (d >= DM) return;
    float acc = 0.f;
    #pragma unroll 8
    for (int s = 0; s < NSL; ++s) acc += slot_init[(size_t)s * SDIM + d];
    sm[d] = acc * (1.0f / NSL);
}

// bias_eff[j] = route_b1[j] + sum_d sm[d] * route_w1[(DM+d)*512 + j]
__global__ __launch_bounds__(256) void spe_biaseff_kernel(const float* __restrict__ sm,
                                                          const float* __restrict__ route_w1,
                                                          const float* __restrict__ route_b1,
                                                          float* __restrict__ be) {
    int j = blockIdx.x * blockDim.x + threadIdx.x;
    if (j >= 512) return;
    float acc = route_b1[j];
    const float* w = route_w1 + (size_t)DM * 512;
    for (int d = 0; d < DM; ++d) acc += sm[d] * w[(size_t)d * 512 + j];
    be[j] = acc;
}

// ---------------------------------------------------------------------------
// Batched WMMA GEMM:  C[z] = act( A[z] @ Bt[z]^T / rowscale + bias )
//   A : [M,K] bf16 row-major,  Bt: [N,K] bf16 row-major (pre-transposed B)
//   template <CBF16 (bf16 out), CTRANS (store C^T), ACT (1=gelu),
//             HASB (bias), HASRS (per-row divide)>
// Block = 128 threads = 4 waves; tile 64x64; K-tile 64 (2 WMMA k-steps);
// double-buffered LDS; staging via async global->LDS (ASYNCcnt) or scalar
// register prefetch fallback. One barrier per K-tile on the async path, two in the fallback.
// ---------------------------------------------------------------------------
template <int CBF16, int CTRANS, int ACT, int HASB, int HASRS>
__global__ __launch_bounds__(128) void spe_gemm_wmma(
    const u16* __restrict__ A, long long lda, long long strideA,
    const u16* __restrict__ Bt, long long ldbt, long long strideB,
    void* __restrict__ C, long long ldc, long long strideC,
    int K,
    const float* __restrict__ bias,
    const float* __restrict__ rowscale, long long rsStride) {
    const int z  = blockIdx.z;
    const int m0 = blockIdx.y * 64;
    const int n0 = blockIdx.x * 64;
    const u16* __restrict__ Ab = A  + (size_t)z * strideA;
    const u16* __restrict__ Bb = Bt + (size_t)z * strideB;

    __shared__ __align__(64) u16 As[2][64 * 64];
    __shared__ __align__(64) u16 Bs[2][64 * 64];

    const int tid  = threadIdx.x;
    const int lane = tid & 31;
    const int w    = tid >> 5;       // wave 0..3
    const int sel  = lane >> 4;      // half-wave select
    const int l16  = lane & 15;

    // 64x64 tile = 512 chunks of 8 bf16 (16B); 4 chunks per thread.
    const int c0 = tid, c1 = tid + 128, c2 = tid + 256, c3 = tid + 384;
    const int r0c = c0 >> 3, a0c = (c0 & 7) * 8;
    const int r1c = c1 >> 3, a1c = (c1 & 7) * 8;
    const int r2c = c2 >> 3, a2c = (c2 & 7) * 8;
    const int r3c = c3 >> 3, a3c = (c3 & 7) * 8;
    // LDS offsets (k XOR-swizzled by row parity); same mapping for A and B.
    const int o0 = r0c * 64 + (a0c ^ ((r0c & 1) << 5));
    const int o1 = r1c * 64 + (a1c ^ ((r1c & 1) << 5));
    const int o2 = r2c * 64 + (a2c ^ ((r2c & 1) << 5));
    const int o3 = r3c * 64 + (a3c ^ ((r3c & 1) << 5));
    const u16* aB0 = Ab + (size_t)(m0 + r0c) * lda + a0c;
    const u16* aB1 = Ab + (size_t)(m0 + r1c) * lda + a1c;
    const u16* aB2 = Ab + (size_t)(m0 + r2c) * lda + a2c;
    const u16* aB3 = Ab + (size_t)(m0 + r3c) * lda + a3c;
    const u16* bB0 = Bb + (size_t)(n0 + r0c) * ldbt + a0c;
    const u16* bB1 = Bb + (size_t)(n0 + r1c) * ldbt + a1c;
    const u16* bB2 = Bb + (size_t)(n0 + r2c) * ldbt + a2c;
    const u16* bB3 = Bb + (size_t)(n0 + r3c) * ldbt + a3c;

    v8f acc[4];
    #pragma unroll
    for (int nf = 0; nf < 4; ++nf)
        #pragma unroll
        for (int j = 0; j < 8; ++j) acc[nf][j] = 0.f;

    const int steps = K >> 6;
    const int arow = 16 * w + l16;
    const int aswz = (arow & 1) << 5;
    const int bswz = (l16 & 1) << 5;

#define SPE_COMPUTE(p)                                                         \
    do {                                                                       \
        _Pragma("unroll")                                                      \
        for (int ks = 0; ks < 64; ks += 32) {                                  \
            v8bf alo = *(const v8bf*)&As[p][arow * 64 + ((ks + 16 * sel) ^ aswz)];       \
            v8bf ahi = *(const v8bf*)&As[p][arow * 64 + ((ks + 8 + 16 * sel) ^ aswz)];   \
            v16bf a;                                                           \
            _Pragma("unroll")                                                  \
            for (int i = 0; i < 8; ++i) { a[i] = alo[i]; a[8 + i] = ahi[i]; }  \
            v16bf bf[4];                                                       \
            _Pragma("unroll")                                                  \
            for (int nf = 0; nf < 4; ++nf)                                     \
                bf[nf] = *(const v16bf*)&Bs[p][(nf * 16 + l16) * 64 + ((ks + 16 * sel) ^ bswz)]; \
            _Pragma("unroll")                                                  \
            for (int nf = 0; nf < 4; ++nf)                                     \
                acc[nf] = __builtin_amdgcn_wmma_f32_16x16x32_bf16(             \
                    false, a, false, bf[nf], (short)0, acc[nf], false, false); \
        }                                                                      \
    } while (0)

#if USE_ASYNC
#define SPE_ISSUE(k0, p)                                                       \
    do {                                                                       \
        async_b128(aB0 + (k0), &As[p][o0]); async_b128(aB1 + (k0), &As[p][o1]);\
        async_b128(aB2 + (k0), &As[p][o2]); async_b128(aB3 + (k0), &As[p][o3]);\
        async_b128(bB0 + (k0), &Bs[p][o0]); async_b128(bB1 + (k0), &Bs[p][o1]);\
        async_b128(bB2 + (k0), &Bs[p][o2]); async_b128(bB3 + (k0), &Bs[p][o3]);\
    } while (0)

    SPE_ISSUE(0, 0);
    for (int it = 0; it < steps; ++it) {
        const int p = it & 1;
        __builtin_amdgcn_s_wait_asynccnt(0);   // own chunks of tile `it` landed
        __syncthreads();                       // everyone's chunks landed
        if (it + 1 < steps) SPE_ISSUE((it + 1) << 6, p ^ 1);  // overlap compute
        SPE_COMPUTE(p);
    }
#undef SPE_ISSUE
#else
    uint4 ra0, ra1, ra2, ra3, rb0, rb1, rb2, rb3;
#define SPE_GLOAD(k0)                                                          \
    do {                                                                       \
        ra0 = *(const uint4*)(aB0 + (k0)); ra1 = *(const uint4*)(aB1 + (k0));  \
        ra2 = *(const uint4*)(aB2 + (k0)); ra3 = *(const uint4*)(aB3 + (k0));  \
        rb0 = *(const uint4*)(bB0 + (k0)); rb1 = *(const uint4*)(bB1 + (k0));  \
        rb2 = *(const uint4*)(bB2 + (k0)); rb3 = *(const uint4*)(bB3 + (k0));  \
    } while (0)

    SPE_GLOAD(0);
    for (int it = 0; it < steps; ++it) {
        const int p = it & 1;
        *(uint4*)&As[p][o0] = ra0; *(uint4*)&As[p][o1] = ra1;
        *(uint4*)&As[p][o2] = ra2; *(uint4*)&As[p][o3] = ra3;
        *(uint4*)&Bs[p][o0] = rb0; *(uint4*)&Bs[p][o1] = rb1;
        *(uint4*)&Bs[p][o2] = rb2; *(uint4*)&Bs[p][o3] = rb3;
        __syncthreads();
        if (it + 1 < steps) SPE_GLOAD((it + 1) << 6);   // overlap with compute
        SPE_COMPUTE(p);
        __syncthreads();
    }
#undef SPE_GLOAD
#endif
#undef SPE_COMPUTE

    // ---- epilogue ----
    const float* rs = HASRS ? rowscale + (size_t)z * rsStride : nullptr;
    #pragma unroll
    for (int nf = 0; nf < 4; ++nf) {
        #pragma unroll
        for (int j = 0; j < 8; ++j) {
            int m = m0 + 16 * w + 8 * sel + j;
            int n = n0 + nf * 16 + l16;
            float v = acc[nf][j];
            if constexpr (HASRS) v /= rs[m];
            if constexpr (HASB)  v += bias[n];
            if constexpr (ACT == 1)
                v = 0.5f * v * (1.0f + erff(v * 0.70710678118f));   // exact gelu
            size_t ci = CTRANS ? (size_t)z * strideC + (size_t)n * ldc + m
                               : (size_t)z * strideC + (size_t)m * ldc + n;
            if constexpr (CBF16) ((u16*)C)[ci]   = f2bf(v);
            else                 ((float*)C)[ci] = v;
        }
    }
}

// ---------------------------------------------------------------------------
// top-k(8) + softmax per row (one wave per row).
// ---------------------------------------------------------------------------
__global__ __launch_bounds__(256) void spe_topk_kernel(const float* __restrict__ logits,
                                                       const float* __restrict__ tau,
                                                       u16* __restrict__ alpha_b,
                                                       u16* __restrict__ alpha_t) {
    int wid  = (blockIdx.x * blockDim.x + threadIdx.x) >> 5;
    int lane = threadIdx.x & 31;
    if (wid >= BT) return;
    const float scale = 1.0f / (fabsf(tau[0]) + 0.1f);
    const float* row = logits + (size_t)wid * NSL;
    float v0 = row[lane] * scale;
    float v1 = row[lane + 32] * scale;

    float kv[KTOP]; int ki[KTOP];
    #pragma unroll
    for (int k = 0; k < KTOP; ++k) {
        float bv; int bi;
        if (v0 >= v1) { bv = v0; bi = lane; } else { bv = v1; bi = lane + 32; }
        #pragma unroll
        for (int off = 16; off; off >>= 1) {
            float ov = __shfl_xor(bv, off);
            int   oi = __shfl_xor(bi, off);
            if (ov > bv || (ov == bv && oi < bi)) { bv = ov; bi = oi; }
        }
        kv[k] = bv; ki[k] = bi;
        if (bi == lane)      v0 = -3.0e38f;
        if (bi == lane + 32) v1 = -3.0e38f;
    }
    float mx = kv[0], s = 0.f, wgt[KTOP];
    #pragma unroll
    for (int k = 0; k < KTOP; ++k) { wgt[k] = expf(kv[k] - mx); s += wgt[k]; }
    float inv = 1.0f / s;

    int b = wid / TSEQ, t = wid % TSEQ;
    for (int ss = lane; ss < NSL; ss += 32) {
        float a = 0.f;
        #pragma unroll
        for (int k = 0; k < KTOP; ++k) if (ki[k] == ss) a = wgt[k] * inv;
        u16 ab = f2bf(a);
        alpha_b[(size_t)wid * NSL + ss] = ab;
        alpha_t[((size_t)b * NSL + ss) * TSEQ + t] = ab;
    }
}

// denom[b,s] = sum_t alpha[b,t,s] + 1e-8
__global__ __launch_bounds__(256) void spe_denom_kernel(const u16* __restrict__ alpha_t,
                                                        float* __restrict__ denom) {
    int row  = (blockIdx.x * blockDim.x + threadIdx.x) >> 5;
    int lane = threadIdx.x & 31;
    if (row >= BATCH * NSL) return;
    const u16* p = alpha_t + (size_t)row * TSEQ;
    float s = 0.f;
    for (int t = lane; t < TSEQ; t += 32) s += bf2f(p[t]);
    #pragma unroll
    for (int off = 16; off; off >>= 1) s += __shfl_xor(s, off);
    if (lane == 0) denom[row] = s + 1e-8f;
}

// ---------------------------------------------------------------------------
// GRU gates: S_new = (1-z)*n + z*S
// ---------------------------------------------------------------------------
__global__ __launch_bounds__(256) void spe_gates_kernel(const float* __restrict__ gi,
                                                        const float* __restrict__ gh,
                                                        const float* __restrict__ slot_init,
                                                        u16* __restrict__ snew) {
    const int NE = BATCH * NSL * SDIM;
    for (int e = blockIdx.x * blockDim.x + threadIdx.x; e < NE;
         e += gridDim.x * blockDim.x) {
        int b = e / (NSL * SDIM);
        int r = e - b * (NSL * SDIM);
        int s = r / SDIM;
        int j = r - s * SDIM;
        size_t gib = ((size_t)b * NSL + s) * (3 * SDIM);
        size_t ghb = (size_t)s * (3 * SDIM);
        float i_r = gi[gib + j],            h_r = gh[ghb + j];
        float i_z = gi[gib + SDIM + j],     h_z = gh[ghb + SDIM + j];
        float i_n = gi[gib + 2 * SDIM + j], h_n = gh[ghb + 2 * SDIM + j];
        float rg = 1.0f / (1.0f + expf(-(i_r + h_r)));
        float zg = 1.0f / (1.0f + expf(-(i_z + h_z)));
        float ng = tanhf(i_n + rg * h_n);
        float S  = slot_init[(size_t)s * SDIM + j];
        snew[e]  = f2bf((1.0f - zg) * ng + zg * S);
    }
}

// ---------------------------------------------------------------------------
// Host side
// ---------------------------------------------------------------------------
template <int CBF16, int CTRANS, int ACT, int HASB, int HASRS>
static void launch_gemm(const u16* A, long long lda, long long sA,
                        const u16* Bt, long long ldbt, long long sB,
                        void* C, long long ldc, long long sC,
                        int M, int N, int K,
                        const float* bias, const float* rs, long long rsStride,
                        int batch, hipStream_t st) {
    dim3 g(N / 64, M / 64, batch);
    spe_gemm_wmma<CBF16, CTRANS, ACT, HASB, HASRS>
        <<<g, 128, 0, st>>>(A, lda, sA, Bt, ldbt, sB, C, ldc, sC, K, bias, rs, rsStride);
}

extern "C" void kernel_launch(void* const* d_in, const int* in_sizes, int n_in,
                              void* d_out, int out_size, void* d_ws, size_t ws_size,
                              hipStream_t stream) {
    (void)in_sizes; (void)n_in; (void)out_size; (void)ws_size;
    const float* x      = (const float*)d_in[0];
    const float* slot0  = (const float*)d_in[1];
    const float* proj_w = (const float*)d_in[2];
    const float* proj_b = (const float*)d_in[3];
    const float* rw1    = (const float*)d_in[4];
    const float* rb1    = (const float*)d_in[5];
    const float* rw2    = (const float*)d_in[6];
    const float* rb2    = (const float*)d_in[7];
    const float* wih    = (const float*)d_in[8];
    const float* whh    = (const float*)d_in[9];
    const float* bih    = (const float*)d_in[10];
    const float* bhh    = (const float*)d_in[11];
    const float* vw     = (const float*)d_in[12];
    const float* vb     = (const float*)d_in[13];
    const float* ow     = (const float*)d_in[14];
    const float* ob     = (const float*)d_in[15];
    const float* tau    = (const float*)d_in[16];
    float* out          = (float*)d_out;

    char* wsp = (char*)d_ws;
    size_t off = 0;
    auto alloc = [&](size_t bytes) -> void* {
        void* p = wsp + off;
        off += (bytes + 255) & ~(size_t)255;
        return p;
    };

    u16*   xb      = (u16*)alloc((size_t)BT * DM * 2);
    u16*   xT      = (u16*)alloc((size_t)BATCH * DM * TSEQ * 2);
    u16*   hb      = (u16*)alloc((size_t)BT * 512 * 2);
    u16*   w1aT    = (u16*)alloc((size_t)512 * DM * 2);
    u16*   w2T     = (u16*)alloc((size_t)NSL * 512 * 2);
    u16*   wpT     = (u16*)alloc((size_t)SDIM * DM * 2);
    u16*   wihT    = (u16*)alloc((size_t)3 * SDIM * SDIM * 2);
    u16*   whhT    = (u16*)alloc((size_t)3 * SDIM * SDIM * 2);
    u16*   vwT     = (u16*)alloc((size_t)DM * SDIM * 2);
    u16*   owT     = (u16*)alloc((size_t)DM * DM * 2);
    u16*   si_b    = (u16*)alloc((size_t)NSL * SDIM * 2);
    float* sm      = (float*)alloc(DM * 4);
    float* be      = (float*)alloc(512 * 4);
    float* logits  = (float*)alloc((size_t)BT * NSL * 4);
    u16*   alpha_b = (u16*)alloc((size_t)BT * NSL * 2);
    u16*   alpha_t = (u16*)alloc((size_t)BATCH * NSL * TSEQ * 2);
    float* denom   = (float*)alloc((size_t)BATCH * NSL * 4);
    u16*   axb     = (u16*)alloc((size_t)BATCH * NSL * DM * 2);
    u16*   slin_b  = (u16*)alloc((size_t)BATCH * NSL * SDIM * 2);
    float* gh      = (float*)alloc((size_t)NSL * 3 * SDIM * 4);
    float* gi      = (float*)alloc((size_t)BATCH * NSL * 3 * SDIM * 4);
    u16*   snew_b  = (u16*)alloc((size_t)BATCH * NSL * SDIM * 2);
    u16*   sv_b    = (u16*)alloc((size_t)BATCH * NSL * DM * 2);
    u16*   sv2T    = (u16*)alloc((size_t)BATCH * DM * NSL * 2);

    auto cvt = [&](const float* s, u16* d, int n) {
        spe_cvt_kernel<<<(n + 255) / 256, 256, 0, stream>>>(s, d, n);
    };
    auto cvtT = [&](const float* s, u16* d, int R, int Ccols, long long sS, long long dS, int z) {
        dim3 g(Ccols / 32, R / 32, z);
        spe_cvtT_kernel<<<g, 256, 0, stream>>>(s, d, R, Ccols, sS, dS);
    };

    cvt(x,     xb,   BT * DM);
    cvt(slot0, si_b, NSL * SDIM);
    cvtT(rw1,    w1aT, DM, 512, 0, 0, 1);                    // top half of route_w1
    cvtT(rw2,    w2T,  512, NSL, 0, 0, 1);
    cvtT(proj_w, wpT,  DM, SDIM, 0, 0, 1);
    cvtT(wih,    wihT, SDIM, 3 * SDIM, 0, 0, 1);
    cvtT(whh,    whhT, SDIM, 3 * SDIM, 0, 0, 1);
    cvtT(vw,     vwT,  SDIM, DM, 0, 0, 1);
    cvtT(ow,     owT,  DM, DM, 0, 0, 1);
    cvtT(x,      xT,   TSEQ, DM, (long long)TSEQ * DM, (long long)DM * TSEQ, BATCH);

    spe_slotmean_kernel<<<(DM + 255) / 256, 256, 0, stream>>>(slot0, sm);
    spe_biaseff_kernel<<<(512 + 255) / 256, 256, 0, stream>>>(sm, rw1, rb1, be);

    // h = gelu(x @ W1a + bias_eff)
    launch_gemm<1, 0, 1, 1, 0>(xb, DM, 0, w1aT, DM, 0, hb, 512, 0,
                               BT, 512, DM, be, nullptr, 0, 1, stream);
    // logits = h @ W2 + b2
    launch_gemm<0, 0, 0, 1, 0>(hb, 512, 0, w2T, 512, 0, logits, NSL, 0,
                               BT, NSL, 512, rb2, nullptr, 0, 1, stream);
    spe_topk_kernel<<<BT / 8, 256, 0, stream>>>(logits, tau, alpha_b, alpha_t);
    spe_denom_kernel<<<(BATCH * NSL) / 8, 256, 0, stream>>>(alpha_t, denom);

    // ax = alpha^T @ x
    launch_gemm<1, 0, 0, 0, 0>(alpha_t, TSEQ, (long long)NSL * TSEQ,
                               xT, TSEQ, (long long)DM * TSEQ,
                               axb, DM, (long long)NSL * DM,
                               NSL, DM, TSEQ, nullptr, nullptr, 0, BATCH, stream);
    // slot_in = ax @ Wp / denom + b_p
    launch_gemm<1, 0, 0, 1, 1>(axb, DM, (long long)NSL * DM,
                               wpT, DM, 0,
                               slin_b, SDIM, (long long)NSL * SDIM,
                               NSL, SDIM, DM, proj_b, denom, NSL, BATCH, stream);
    // gh = slot_init @ W_hh + b_hh
    launch_gemm<0, 0, 0, 1, 0>(si_b, SDIM, 0, whhT, SDIM, 0,
                               gh, 3 * SDIM, 0,
                               NSL, 3 * SDIM, SDIM, bhh, nullptr, 0, 1, stream);
    // gi = slot_in @ W_ih + b_ih
    launch_gemm<0, 0, 0, 1, 0>(slin_b, SDIM, (long long)NSL * SDIM,
                               wihT, SDIM, 0,
                               gi, 3 * SDIM, (long long)NSL * 3 * SDIM,
                               NSL, 3 * SDIM, SDIM, bih, nullptr, 0, BATCH, stream);
    spe_gates_kernel<<<(BATCH * NSL * SDIM) / 256, 256, 0, stream>>>(gi, gh, slot0, snew_b);

    // sv = S_new @ value_w + value_b
    launch_gemm<1, 0, 0, 1, 0>(snew_b, SDIM, (long long)NSL * SDIM,
                               vwT, SDIM, 0,
                               sv_b, DM, (long long)NSL * DM,
                               NSL, DM, SDIM, vb, nullptr, 0, BATCH, stream);
    // sv2^T = (sv @ out_w)^T  (stored transposed for the combine GEMM)
    launch_gemm<1, 1, 0, 0, 0>(sv_b, DM, (long long)NSL * DM,
                               owT, DM, 0,
                               sv2T, NSL, (long long)DM * NSL,
                               NSL, DM, DM, nullptr, nullptr, 0, BATCH, stream);
    // out = alpha @ sv2 + out_b
    launch_gemm<0, 0, 0, 1, 0>(alpha_b, NSL, (long long)TSEQ * NSL,
                               sv2T, NSL, (long long)DM * NSL,
                               out, DM, (long long)TSEQ * DM,
                               TSEQ, DM, NSL, ob, nullptr, 0, BATCH, stream);
}
